// NPerTokenSwishGLU_64613488001346
// MI455X (gfx1250) — matrix operand order repaired and checked by reading, not compiled, not run on hardware
//
#include <hip/hip_runtime.h>
#include <hip/hip_bf16.h>
#include <math.h>

// Problem sizes (compile-time)
#define BB 64
#define LL 128
#define DD 512
#define FF 1024
#define EPSV 1e-12f
#define PITCH 20   // dwords per LDS row: 80B -> 16B aligned for b128, conflict-free bank stride

typedef _Float16 f16_t;
typedef __attribute__((ext_vector_type(16))) _Float16 v16h;
typedef __attribute__((ext_vector_type(8)))  float    v8f;
typedef __attribute__((ext_vector_type(4)))  unsigned v4u;

union Op16 { v4u q[2]; v16h m; };

__device__ __forceinline__ unsigned pkh(float a, float b) {
    union { _Float16 h[2]; unsigned u; } p;
    p.h[0] = (_Float16)a; p.h[1] = (_Float16)b;
    return p.u;
}

__device__ __forceinline__ v8f wmma16(const Op16& a, const Op16& b, v8f c) {
    // D = A(16x32 f16) * B(32x16 f16) + C(16x16 f32)
    return __builtin_amdgcn_wmma_f32_16x16x32_f16(false, a.m, false, b.m,
                                                  (short)0, c, false, false);
}

// ---------------------------------------------------------------------------
// Phase 1: u = x @ Wg, v = x @ Wu (per l), column-norm post-scale, SiLU-gate,
//          write h (f16) to workspace.  Grid: (F/64, L), 256 threads.
// ---------------------------------------------------------------------------
__global__ __launch_bounds__(256)
void ngpt_gateup_kernel(const float* __restrict__ x,
                        const float* __restrict__ Wg,
                        const float* __restrict__ Wu,
                        const float* __restrict__ su,
                        const float* __restrict__ sv,
                        f16_t* __restrict__ hws) {
    __shared__ unsigned sA[64 * PITCH];   // x tile   [b][kpair]
    __shared__ unsigned sG[64 * PITCH];   // Wg tile  [f'][kpair]
    __shared__ unsigned sU[64 * PITCH];   // Wu tile  [f'][kpair]
    __shared__ float nsG[64];
    __shared__ float nsU[64];

    const int t    = threadIdx.x;
    const int lane = t & 31;
    const int wave = t >> 5;
    const int l    = blockIdx.y;
    const int f0   = blockIdx.x * 64;

    if (t < 64) { nsG[t] = 0.f; nsU[t] = 0.f; }

    // staging roles
    const int bS = t >> 2, kq = t & 3;          // x: row b, quarter of K
    const int fg = (t & 15) * 4, rS = t >> 4;   // W: 4 f-cols, k-pair row

    // compute roles
    const int hi = lane >> 4, lm = lane & 15;
    const int mt = wave & 3;
    const int nt0 = (wave >> 2) * 2, nt1 = nt0 + 1;

    float4 ag = {0.f,0.f,0.f,0.f}, au = {0.f,0.f,0.f,0.f};
    v8f cu0 = {}, cu1 = {}, cv0 = {}, cv1 = {};

    const float* xp = x  + ((size_t)bS * LL + l) * DD + kq * 8;
    const float* gp = Wg + ((size_t)l * DD + 2 * rS) * FF + f0 + fg;
    const float* up = Wu + ((size_t)l * DD + 2 * rS) * FF + f0 + fg;

    for (int ks = 0; ks < DD / 32; ++ks) {
        const int k0 = ks * 32;
        // ---- global loads (fp32, coalesced b128) ----
        float4 x0 = *(const float4*)(xp + k0);
        float4 x1 = *(const float4*)(xp + k0 + 4);
        const float* g0 = gp + (size_t)k0 * FF;
        float4 wg0 = *(const float4*)(g0);
        float4 wg1 = *(const float4*)(g0 + FF);
        const float* u0 = up + (size_t)k0 * FF;
        float4 wu0 = *(const float4*)(u0);
        float4 wu1 = *(const float4*)(u0 + FF);
        // prefetch next K-step's weight rows (global_prefetch_b8)
        if (ks + 1 < DD / 32) {
            __builtin_prefetch(g0 + (size_t)32 * FF, 0, 1);
            __builtin_prefetch(u0 + (size_t)32 * FF, 0, 1);
        }
        // column sum-of-squares (normalization commutes with the GEMM)
        ag.x += wg0.x*wg0.x + wg1.x*wg1.x;  ag.y += wg0.y*wg0.y + wg1.y*wg1.y;
        ag.z += wg0.z*wg0.z + wg1.z*wg1.z;  ag.w += wg0.w*wg0.w + wg1.w*wg1.w;
        au.x += wu0.x*wu0.x + wu1.x*wu1.x;  au.y += wu0.y*wu0.y + wu1.y*wu1.y;
        au.z += wu0.z*wu0.z + wu1.z*wu1.z;  au.w += wu0.w*wu0.w + wu1.w*wu1.w;

        __syncthreads();   // prior iteration's LDS reads complete
        // ---- stage to LDS as packed f16 k-pairs ----
        {
            v4u pa = { pkh(x0.x, x0.y), pkh(x0.z, x0.w),
                       pkh(x1.x, x1.y), pkh(x1.z, x1.w) };
            *(v4u*)&sA[bS * PITCH + kq * 4] = pa;
        }
        sG[(fg + 0) * PITCH + rS] = pkh(wg0.x, wg1.x);
        sG[(fg + 1) * PITCH + rS] = pkh(wg0.y, wg1.y);
        sG[(fg + 2) * PITCH + rS] = pkh(wg0.z, wg1.z);
        sG[(fg + 3) * PITCH + rS] = pkh(wg0.w, wg1.w);
        sU[(fg + 0) * PITCH + rS] = pkh(wu0.x, wu1.x);
        sU[(fg + 1) * PITCH + rS] = pkh(wu0.y, wu1.y);
        sU[(fg + 2) * PITCH + rS] = pkh(wu0.z, wu1.z);
        sU[(fg + 3) * PITCH + rS] = pkh(wu0.w, wu1.w);
        __syncthreads();

        // ---- WMMA operand fetch (contiguous k = 16*half + s, same order for A and B) ----
        Op16 a;
        {
            const int ab = (mt * 16 + lm) * PITCH + hi * 8;
            a.q[0] = *(const v4u*)&sA[ab];        // K 0-7  (lanes>=16: 16-23)
            a.q[1] = *(const v4u*)&sA[ab + 4];    // K 8-15 (lanes>=16: 24-31)
        }
        const int b0 = (nt0 * 16 + lm) * PITCH + hi * 8;
        const int b1 = (nt1 * 16 + lm) * PITCH + hi * 8;
        Op16 bg0, bg1, bu0, bu1;
        bg0.q[0] = *(const v4u*)&sG[b0];  bg0.q[1] = *(const v4u*)&sG[b0 + 4];
        bg1.q[0] = *(const v4u*)&sG[b1];  bg1.q[1] = *(const v4u*)&sG[b1 + 4];
        bu0.q[0] = *(const v4u*)&sU[b0];  bu0.q[1] = *(const v4u*)&sU[b0 + 4];
        bu1.q[0] = *(const v4u*)&sU[b1];  bu1.q[1] = *(const v4u*)&sU[b1 + 4];

        cu0 = wmma16(a, bg0, cu0);
        cv0 = wmma16(a, bu0, cv0);
        cu1 = wmma16(a, bg1, cu1);
        cv1 = wmma16(a, bu1, cv1);
    }

    // reduce column sumsq
    atomicAdd(&nsG[fg + 0], ag.x); atomicAdd(&nsG[fg + 1], ag.y);
    atomicAdd(&nsG[fg + 2], ag.z); atomicAdd(&nsG[fg + 3], ag.w);
    atomicAdd(&nsU[fg + 0], au.x); atomicAdd(&nsU[fg + 1], au.y);
    atomicAdd(&nsU[fg + 2], au.z); atomicAdd(&nsU[fg + 3], au.w);
    __syncthreads();

    // ---- epilogue: norm post-scale + NGPT scales + SiLU gate -> h (f16) ----
    const float vsc = 22.62741699796952f;   // sqrt(D)
    #pragma unroll
    for (int half = 0; half < 2; ++half) {
        const int nt = half ? nt1 : nt0;
        const v8f cu = half ? cu1 : cu0;
        const v8f cv = half ? cv1 : cv0;
        const int fc = nt * 16 + lm;
        const float uscale = fabsf(su[f0 + fc]) / fmaxf(sqrtf(nsG[fc]), EPSV);
        const float vscale = fabsf(sv[f0 + fc]) * vsc / fmaxf(sqrtf(nsU[fc]), EPSV);
        #pragma unroll
        for (int r = 0; r < 8; ++r) {
            const int brow = mt * 16 + hi * 8 + r;
            const float uu = cu[r] * uscale;
            const float vv = cv[r] * vscale;
            const float hv = (vv / (1.f + __expf(-vv))) * uu;   // silu(vv)*uu
            hws[((size_t)brow * LL + l) * FF + f0 + fc] = (f16_t)hv;
        }
    }
}

// ---------------------------------------------------------------------------
// Phase 2: out = h @ Wd (per l) with Wd column-norm post-scale.
//          h tile (already f16) is DMA'd straight into LDS with
//          global_load_async_to_lds_b128 (ASYNCcnt path); Wd goes through
//          registers for fp32->f16 conversion.  Grid: (D/64, L), 256 threads.
// ---------------------------------------------------------------------------
__global__ __launch_bounds__(256)
void ngpt_down_kernel(const f16_t* __restrict__ hws,
                      const float* __restrict__ Wd,
                      float* __restrict__ out) {
    __shared__ unsigned sA[64 * PITCH];   // h tile  [b][fpair]  (async-filled)
    __shared__ unsigned sB[64 * PITCH];   // Wd tile [d'][fpair]
    __shared__ float nsD[64];

    const int t    = threadIdx.x;
    const int lane = t & 31;
    const int wave = t >> 5;
    const int l    = blockIdx.y;
    const int d0   = blockIdx.x * 64;

    if (t < 64) nsD[t] = 0.f;

    const int bS = t >> 2, q = t & 3;
    const int dg = (t & 15) * 4, rS = t >> 4;
    const int hi = lane >> 4, lm = lane & 15;
    const int mt = wave & 3;
    const int nt0 = (wave >> 2) * 2, nt1 = nt0 + 1;

    float4 ad = {0.f,0.f,0.f,0.f};
    v8f c0 = {}, c1 = {};

    const f16_t* hp = hws + ((size_t)bS * LL + l) * FF + q * 8;
    const float* wp = Wd + ((size_t)l * FF + 2 * rS) * DD + d0 + dg;
    // LDS byte offset for this thread's async destination (low 32 bits of
    // the generic shared-memory address are the DS byte offset)
    const unsigned ldsA = (unsigned)(uintptr_t)&sA[bS * PITCH + q * 4];

    for (int ks = 0; ks < FF / 32; ++ks) {
        const int k0 = ks * 32;
        // ---- Wd: fp32 global loads for conversion path ----
        const float* w0p = wp + (size_t)k0 * DD;
        float4 w0 = *(const float4*)(w0p);
        float4 w1 = *(const float4*)(w0p + DD);
        if (ks + 1 < FF / 32) {
            __builtin_prefetch(w0p + (size_t)32 * DD, 0, 1);
        }
        ad.x += w0.x*w0.x + w1.x*w1.x;  ad.y += w0.y*w0.y + w1.y*w1.y;
        ad.z += w0.z*w0.z + w1.z*w1.z;  ad.w += w0.w*w0.w + w1.w*w1.w;

        __syncthreads();   // previous iteration's LDS reads retired
        // ---- h tile: async DMA straight into LDS (no VGPR round trip) ----
        {
            const f16_t* hsrc = hp + k0;
            asm volatile("global_load_async_to_lds_b128 %0, %1, off"
                         :: "v"(ldsA), "v"(hsrc)
                         : "memory");
        }
        // ---- Wd tile: pack f16 k-pairs into LDS ----
        sB[(dg + 0) * PITCH + rS] = pkh(w0.x, w1.x);
        sB[(dg + 1) * PITCH + rS] = pkh(w0.y, w1.y);
        sB[(dg + 2) * PITCH + rS] = pkh(w0.z, w1.z);
        sB[(dg + 3) * PITCH + rS] = pkh(w0.w, w1.w);
        asm volatile("s_wait_asynccnt 0x0" ::: "memory");
        __syncthreads();

        Op16 a;
        {
            const int ab = (mt * 16 + lm) * PITCH + hi * 8;
            a.q[0] = *(const v4u*)&sA[ab];
            a.q[1] = *(const v4u*)&sA[ab + 4];
        }
        const int b0 = (nt0 * 16 + lm) * PITCH + hi * 8;
        const int b1 = (nt1 * 16 + lm) * PITCH + hi * 8;
        Op16 bb0, bb1;
        bb0.q[0] = *(const v4u*)&sB[b0];  bb0.q[1] = *(const v4u*)&sB[b0 + 4];
        bb1.q[0] = *(const v4u*)&sB[b1];  bb1.q[1] = *(const v4u*)&sB[b1 + 4];

        c0 = wmma16(a, bb0, c0);
        c1 = wmma16(a, bb1, c1);
    }

    atomicAdd(&nsD[dg + 0], ad.x); atomicAdd(&nsD[dg + 1], ad.y);
    atomicAdd(&nsD[dg + 2], ad.z); atomicAdd(&nsD[dg + 3], ad.w);
    __syncthreads();

    #pragma unroll
    for (int half = 0; half < 2; ++half) {
        const int nt = half ? nt1 : nt0;
        const v8f c = half ? c1 : c0;
        const int dc = nt * 16 + lm;
        const float inv = 1.f / fmaxf(sqrtf(nsD[dc]), EPSV);
        #pragma unroll
        for (int r = 0; r < 8; ++r) {
            const int brow = mt * 16 + hi * 8 + r;
            out[((size_t)brow * LL + l) * DD + d0 + dc] = c[r] * inv;
        }
    }
}

// ---------------------------------------------------------------------------
// Phase 3: per-token L2 row normalization of out (512 floats per row).
// ---------------------------------------------------------------------------
__global__ __launch_bounds__(256)
void ngpt_rownorm_kernel(float* __restrict__ out) {
    __shared__ float red[256];
    const int t = threadIdx.x;
    float* p = out + (size_t)blockIdx.x * DD;
    float a = p[t], b = p[t + 256];
    red[t] = a * a + b * b;
    __syncthreads();
    #pragma unroll
    for (int s = 128; s > 0; s >>= 1) {
        if (t < s) red[t] += red[t + s];
        __syncthreads();
    }
    const float inv = 1.f / fmaxf(sqrtf(red[0]), EPSV);
    p[t]       = a * inv;
    p[t + 256] = b * inv;
}

// ---------------------------------------------------------------------------
extern "C" void kernel_launch(void* const* d_in, const int* in_sizes, int n_in,
                              void* d_out, int out_size, void* d_ws, size_t ws_size,
                              hipStream_t stream) {
    (void)in_sizes; (void)n_in; (void)out_size; (void)ws_size;
    const float* x  = (const float*)d_in[0];
    const float* Wg = (const float*)d_in[1];
    const float* Wu = (const float*)d_in[2];
    const float* Wd = (const float*)d_in[3];
    const float* su = (const float*)d_in[4];
    const float* sv = (const float*)d_in[5];
    float* out = (float*)d_out;
    f16_t* hws = (f16_t*)d_ws;   // B*L*F f16 = 16.8 MB

    ngpt_gateup_kernel<<<dim3(FF / 64, LL), 256, 0, stream>>>(x, Wg, Wu, su, sv, hws);
    ngpt_down_kernel  <<<dim3(DD / 64, LL), 256, 0, stream>>>(hws, Wd, out);
    ngpt_rownorm_kernel<<<BB * LL, 256, 0, stream>>>(out);
}
